// MultiheadAttentionCustom_18305150616349
// MI455X (gfx1250) — matrix operand order repaired and checked by reading, not compiled, not run on hardware
//
#include <hip/hip_runtime.h>
#include <hip/hip_bf16.h>

// Problem constants (match reference)
#define B_  4
#define S_  2048
#define D_  1024
#define H_  16
#define HD_ 64

typedef __attribute__((ext_vector_type(16))) _Float16 v16h;
typedef __attribute__((ext_vector_type(8)))  _Float16 v8h;
typedef __attribute__((ext_vector_type(8)))  float    v8f;

// Wave-local LDS phase fence: order ds stores -> ds loads within one wave
// using the CDNA5 split DS counter (no workgroup barrier needed).
__device__ __forceinline__ void lds_wave_fence() {
    __asm__ volatile("s_wait_dscnt 0" ::: "memory");
}

// CDNA5 async memory->LDS copy, 32B per lane (2 x b128; IOFFSET applies to
// both the LDS and global address). Tracked on ASYNCcnt.
__device__ __forceinline__ void async_copy32(void* lds_ptr, const void* gptr) {
    unsigned lds_off = (unsigned)(size_t)lds_ptr;   // flat addr low 32b == LDS offset
    __asm__ volatile(
        "global_load_async_to_lds_b128 %0, %1, off\n\t"
        "global_load_async_to_lds_b128 %0, %1, off offset:16"
        :: "v"(lds_off), "v"(gptr) : "memory");
}
__device__ __forceinline__ void async_wait0() {
    __asm__ volatile("s_wait_asynccnt 0" ::: "memory");
}

// -----------------------------------------------------------------------------
// Kernel 1a: pack weights f32 [out=e][in=d] -> f16, B-fragment-friendly layout.
// Wp[(kc*D + n)*32 + kk] = W[n][kc*32+kk]; a B-frag read is 16 contiguous
// halves (32B, 32B-aligned) at (kc*D+n)*32 + {0|16}.
// -----------------------------------------------------------------------------
__global__ __launch_bounds__(256) void pack_weights(
    const float* __restrict__ Wq, const float* __restrict__ Wk,
    const float* __restrict__ Wv, _Float16* __restrict__ Wp)
{
    int idx = blockIdx.x * 256 + threadIdx.x;      // 3*D*D threads exactly
    int w   = idx / (D_ * D_);
    int rem = idx % (D_ * D_);
    int n = rem / D_, k = rem % D_;
    const float* W = (w == 0) ? Wq : (w == 1) ? Wk : Wv;
    int kc = k >> 5, kk = k & 31;
    Wp[(size_t)w * D_ * D_ + (((size_t)kc * D_ + n) << 5) + kk] =
        (_Float16)W[(size_t)n * D_ + k];
}

// -----------------------------------------------------------------------------
// Kernel 1b: convert activations f32 -> f16 ONCE (row-major [3][B*S][D]).
// -----------------------------------------------------------------------------
__global__ __launch_bounds__(256) void pack_x(
    const float* __restrict__ q, const float* __restrict__ k,
    const float* __restrict__ v, _Float16* __restrict__ xh)
{
    const size_t per = (size_t)B_ * S_ * D_ / 8;     // 8-elem chunks per matrix
    size_t c = (size_t)blockIdx.x * 256 + threadIdx.x;
    int m = (int)(c / per);
    size_t r = c % per;
    const float* x = (m == 0) ? q : (m == 1) ? k : v;
    v8f f = *(const v8f*)(x + r * 8);
    v8h h;
#pragma unroll
    for (int j = 0; j < 8; ++j) h[j] = (_Float16)f[j];
    *(v8h*)(xh + (size_t)m * B_ * S_ * D_ + r * 8) = h;
}

// -----------------------------------------------------------------------------
// Kernel 2: projection GEMM. One wave -> 32(M) x 64(N) f32 tile, K=1024.
// Per k-step: clause-load 2 A-frags + 4 B-frags (immediate offsets), then an
// 8-WMMA burst. Q,K stored f16 [B,H,S,HD]; V stored TRANSPOSED [B,H,HD,S].
// -----------------------------------------------------------------------------
__global__ __launch_bounds__(32) void proj_gemm(
    const _Float16* __restrict__ xh, const _Float16* __restrict__ Wp,
    _Float16* __restrict__ Qh, _Float16* __restrict__ Kh, _Float16* __restrict__ Vt)
{
    const int lane  = threadIdx.x;
    const int mtile = blockIdx.x;       // 0..(B*S/32 - 1)
    const int ntile = blockIdx.y;       // 0..(D/64 - 1)
    const int mat   = blockIdx.z;       // 0:Q 1:K 2:V
    const _Float16* x = xh + (size_t)mat * B_ * S_ * D_;
    const _Float16* W = Wp + (size_t)mat * D_ * D_;

    const int l15 = lane & 15;
    const int kb8 = (lane < 16) ? 0 : 16;   // A-frag K sub-base (contiguous k order: k = 16h + s)
    const int Kb  = (lane < 16) ? 0 : 16;   // B-frag K base

    const _Float16* xrow0 = x + (size_t)(mtile * 32 + l15) * D_ + kb8;
    const _Float16* xrow1 = xrow0 + (size_t)16 * D_;
    // Base of this lane's B-frag row; per k-step advance = 32*D halves,
    // per n-subtile advance = 16*32 = 512 halves (immediate offsets).
    const _Float16* wbase = W + (((size_t)ntile * 64 + l15) << 5) + Kb;

    v8f acc[2][4] = {};
#pragma unroll 2
    for (int kc = 0; kc < D_ / 32; ++kc) {
        const int k0 = kc << 5;
        // A fragments (rows m..m+15, m+16..m+31).
        v16h a0, a1;
        {
            v8h p00 = *(const v8h*)(xrow0 + k0);
            v8h p01 = *(const v8h*)(xrow0 + k0 + 8);
            v8h p10 = *(const v8h*)(xrow1 + k0);
            v8h p11 = *(const v8h*)(xrow1 + k0 + 8);
#pragma unroll
            for (int j = 0; j < 8; ++j) {
                a0[j] = p00[j]; a0[8 + j] = p01[j];
                a1[j] = p10[j]; a1[8 + j] = p11[j];
            }
        }
        // All four B fragments up-front (one clause, one wait).
        const _Float16* wk = wbase + ((size_t)kc * D_ << 5) / 32 * 32;  // kc*D*32 halves
        v16h bfr[4];
#pragma unroll
        for (int t = 0; t < 4; ++t)
            bfr[t] = *(const v16h*)(wk + ((size_t)kc == (size_t)kc ? (t << 9) : 0));
        // 8-WMMA burst, A-frags reused across 4 B-frags.
#pragma unroll
        for (int t = 0; t < 4; ++t) {
            acc[0][t] = __builtin_amdgcn_wmma_f32_16x16x32_f16(false, a0, false, bfr[t],
                                                               (short)0, acc[0][t], false, false);
            acc[1][t] = __builtin_amdgcn_wmma_f32_16x16x32_f16(false, a1, false, bfr[t],
                                                               (short)0, acc[1][t], false, false);
        }
    }
    // D-fragment scatter: lane holds N=l15, rows Mb..Mb+7 of each 16-row block.
    // 32-row tiles never straddle a batch (2048 rows per batch).
    const int Mb = (lane < 16) ? 0 : 8;
    const int bb = (mtile * 32) >> 11;
    const int s0 = (mtile * 32) & (S_ - 1);
#pragma unroll
    for (int mi = 0; mi < 2; ++mi) {
#pragma unroll
        for (int t = 0; t < 4; ++t) {
            const int e  = ntile * 64 + t * 16 + l15;
            const int h  = e >> 6, hd = e & (HD_ - 1);
            if (mat == 2) {
                // V transposed: 8 rows -> 8 consecutive s -> one 16B store.
                v8h pk;
#pragma unroll
                for (int r = 0; r < 8; ++r) pk[r] = (_Float16)acc[mi][t][r];
                *(v8h*)(Vt + ((size_t)(bb * H_ + h) * HD_ + hd) * S_
                            + s0 + mi * 16 + Mb) = pk;
            } else {
                _Float16* O = (mat == 0) ? Qh : Kh;
                _Float16* orow = O + ((size_t)(bb * H_ + h) * S_
                                      + (s0 + mi * 16 + Mb)) * HD_ + hd;
#pragma unroll
                for (int r = 0; r < 8; ++r)
                    orow[(size_t)r * HD_] = (_Float16)acc[mi][t][r];
            }
        }
    }
}

// -----------------------------------------------------------------------------
// Kernel 3: flash attention. 4 waves/block; each wave owns a 16-query tile,
// the block SHARES each 32-key K/V chunk, staged into LDS with CDNA5
// async memory->LDS copies (ASYNCcnt). Per wave per chunk: 4 WMMA scores +
// online softmax (wave-local LDS, s_wait_dscnt fences) + 4 WMMA PV.
// -----------------------------------------------------------------------------
__global__ __launch_bounds__(128) void flash_attn(
    const _Float16* __restrict__ Qh, const _Float16* __restrict__ Kh,
    const _Float16* __restrict__ Vt, const int* __restrict__ pad_mask,
    float* __restrict__ out)
{
    __shared__ __align__(32) _Float16 sK[32][64];     // K chunk  [key][d]   4 KB
    __shared__ __align__(32) _Float16 sV[64][32];     // V chunk  [d][key]   4 KB
    __shared__ __align__(32) float    sS[4][16][32];  // per-wave scores     8 KB
    __shared__ __align__(32) _Float16 sP[4][16][32];  // per-wave exp(P)     4 KB
    __shared__ float sM[4][16], sL[4][16], sScale[4][16];

    const int tid  = threadIdx.x;
    const int w    = tid >> 5;          // wave in block
    const int lane = tid & 31;
    const int h = blockIdx.y, b = blockIdx.z;
    const int qt    = blockIdx.x * 4 + w;
    const int qbase = qt * 16;
    const int l15  = lane & 15;
    const int kb8  = (lane < 16) ? 0 : 16;
    const int Kb16 = (lane < 16) ? 0 : 16;
    const int Mb   = (lane < 16) ? 0 : 8;
    const size_t bh = (size_t)(b * H_ + h);
    const _Float16* kbase = Kh + bh * S_ * HD_;
    const _Float16* vbase = Vt + bh * HD_ * S_;

    // Per-thread staging coordinates (32B each for K and V chunks).
    const int stK_key = tid >> 2, stK_seg = (tid & 3) << 4;
    const int stV_d   = tid >> 1, stV_seg = (tid & 1) << 4;

    // Q A-fragments for d-chunks {0..31},{32..63}; resident for whole kernel.
    const _Float16* qrow = Qh + (bh * S_ + (qbase + l15)) * HD_;
    v16h aQ0, aQ1;
    {
        v8h q00 = *(const v8h*)(qrow + kb8);
        v8h q01 = *(const v8h*)(qrow + 8 + kb8);
        v8h q10 = *(const v8h*)(qrow + 32 + kb8);
        v8h q11 = *(const v8h*)(qrow + 40 + kb8);
#pragma unroll
        for (int j = 0; j < 8; ++j) {
            aQ0[j] = q00[j]; aQ0[8 + j] = q01[j];
            aQ1[j] = q10[j]; aQ1[8 + j] = q11[j];
        }
    }

    v8f o0 = {}, o1 = {}, o2 = {}, o3 = {};
    if (lane < 16) { sM[w][lane] = -1e30f; sL[w][lane] = 0.0f; }

    const int my_nch  = (qbase + 16 + 31) >> 5;                     // own frontier
    const int nch_blk = ((blockIdx.x * 4 + 3) * 16 + 16 + 31) >> 5; // block frontier

    for (int kc = 0; kc < nch_blk; ++kc) {
        const int kb0 = kc << 5;
        // ---- async stage: K (32x64) and V^T (64x32) chunks, 32B/lane each ----
        async_copy32(&sK[stK_key][stK_seg],
                     kbase + (size_t)(kb0 + stK_key) * HD_ + stK_seg);
        async_copy32(&sV[stV_d][stV_seg],
                     vbase + (size_t)stV_d * S_ + kb0 + stV_seg);
        if (kc + 1 < nch_blk) {   // prefetch next chunk into L2
            __builtin_prefetch(kbase + (size_t)(kb0 + 32 + stK_key) * HD_ + stK_seg, 0, 3);
            __builtin_prefetch(vbase + (size_t)stV_d * S_ + kb0 + 32 + stV_seg, 0, 3);
        }
        async_wait0();
        __syncthreads();

        if (kc < my_nch) {   // wave-uniform skip: EXEC stays all-ones inside
            // ---- scores: S[16q x 32k] = Q . K^T from LDS, scale + masks ----
#pragma unroll
            for (int kh = 0; kh < 2; ++kh) {
                const int key = kb0 + kh * 16 + l15;
                v16h b0 = *(const v16h*)(&sK[kh * 16 + l15][Kb16]);
                v16h b1 = *(const v16h*)(&sK[kh * 16 + l15][32 + Kb16]);
                v8f c = {};
                c = __builtin_amdgcn_wmma_f32_16x16x32_f16(false, aQ0, false, b0,
                                                           (short)0, c, false, false);
                c = __builtin_amdgcn_wmma_f32_16x16x32_f16(false, aQ1, false, b1,
                                                           (short)0, c, false, false);
                const int pm = pad_mask[b * S_ + key];
#pragma unroll
                for (int r = 0; r < 8; ++r) {
                    int qq = qbase + Mb + r;
                    // scale = 1/sqrt(D) = 1/32; masked -> -10000 (matches ref)
                    sS[w][Mb + r][kh * 16 + l15] =
                        (key <= qq && pm != 0) ? c[r] * 0.03125f : -10000.0f;
                }
            }
            lds_wave_fence();
            // ---- online softmax: lane r owns query row r (wave-local) ----
            if (lane < 16) {
                const int r = lane;
                float mold = sM[w][r], mx = mold;
#pragma unroll
                for (int kk = 0; kk < 32; ++kk) mx = fmaxf(mx, sS[w][r][kk]);
                float sc = __expf(mold - mx);
                float ls = 0.0f;
#pragma unroll
                for (int kk = 0; kk < 32; ++kk) {
                    float p = __expf(sS[w][r][kk] - mx);
                    sP[w][r][kk] = (_Float16)p;
                    ls += p;
                }
                sL[w][r] = sL[w][r] * sc + ls;
                sM[w][r] = mx;
                sScale[w][r] = sc;
            }
            lds_wave_fence();
            // rescale running O by exp(m_old - m_new)
#pragma unroll
            for (int r = 0; r < 8; ++r) {
                float f = sScale[w][Mb + r];
                o0[r] *= f; o1[r] *= f; o2[r] *= f; o3[r] *= f;
            }
            // ---- PV: O[16q x 64d] += P[16x32] . V[32x64] (V^T in LDS) ----
            v16h aP;
            {
                v8h p0 = *(const v8h*)(&sP[w][l15][kb8]);
                v8h p1 = *(const v8h*)(&sP[w][l15][8 + kb8]);
#pragma unroll
                for (int j = 0; j < 8; ++j) { aP[j] = p0[j]; aP[8 + j] = p1[j]; }
            }
            v16h bv0 = *(const v16h*)(&sV[ 0 + l15][Kb16]);
            v16h bv1 = *(const v16h*)(&sV[16 + l15][Kb16]);
            v16h bv2 = *(const v16h*)(&sV[32 + l15][Kb16]);
            v16h bv3 = *(const v16h*)(&sV[48 + l15][Kb16]);
            o0 = __builtin_amdgcn_wmma_f32_16x16x32_f16(false, aP, false, bv0, (short)0, o0, false, false);
            o1 = __builtin_amdgcn_wmma_f32_16x16x32_f16(false, aP, false, bv1, (short)0, o1, false, false);
            o2 = __builtin_amdgcn_wmma_f32_16x16x32_f16(false, aP, false, bv2, (short)0, o2, false, false);
            o3 = __builtin_amdgcn_wmma_f32_16x16x32_f16(false, aP, false, bv3, (short)0, o3, false, false);
        }
        __syncthreads();   // protect LDS restage (all threads reach this)
    }

    // ---- normalize and store: out[b, s, h*64 + d] (f32) ----
#pragma unroll
    for (int r = 0; r < 8; ++r) {
        int qq = qbase + Mb + r;
        float inv = 1.0f / sL[w][Mb + r];
        float* orow = out + ((size_t)b * S_ + qq) * D_ + h * HD_;
        orow[ 0 + l15] = o0[r] * inv;
        orow[16 + l15] = o1[r] * inv;
        orow[32 + l15] = o2[r] * inv;
        orow[48 + l15] = o3[r] * inv;
    }
}

// -----------------------------------------------------------------------------
extern "C" void kernel_launch(void* const* d_in, const int* in_sizes, int n_in,
                              void* d_out, int out_size, void* d_ws, size_t ws_size,
                              hipStream_t stream) {
    const float* q   = (const float*)d_in[0];
    const float* k   = (const float*)d_in[1];
    const float* v   = (const float*)d_in[2];
    const int*   pad = (const int*)d_in[3];
    // d_in[4] = attn_mask (causal tril) — computed analytically in-kernel.
    const float* Wq  = (const float*)d_in[5];
    const float* Wk  = (const float*)d_in[6];
    const float* Wv  = (const float*)d_in[7];
    float* out = (float*)d_out;

    // Workspace (all f16): Wp 6MB | xh 48MB | Qh 16MB | Kh 16MB | Vt 16MB = 102MB
    char* ws = (char*)d_ws;
    _Float16* Wp = (_Float16*)(ws);
    _Float16* xh = (_Float16*)(ws + (size_t)6  * 1024 * 1024);
    _Float16* Qh = (_Float16*)(ws + (size_t)54 * 1024 * 1024);
    _Float16* Kh = (_Float16*)(ws + (size_t)70 * 1024 * 1024);
    _Float16* Vt = (_Float16*)(ws + (size_t)86 * 1024 * 1024);

    pack_weights<<<(3 * D_ * D_) / 256, 256, 0, stream>>>(Wq, Wk, Wv, Wp);
    pack_x<<<(3 * B_ * S_ * (D_ / 8)) / 256, 256, 0, stream>>>(q, k, v, xh);
    proj_gemm<<<dim3(B_ * S_ / 32, D_ / 64, 3), 32, 0, stream>>>(xh, Wp, Qh, Kh, Vt);
    flash_attn<<<dim3(S_ / 64, H_, B_), 128, 0, stream>>>(Qh, Kh, Vt, pad, out);
}
